// RelativePositionEncoder_39041252721256
// MI455X (gfx1250) — compile-verified, matrix operand order checked by reading, not run on hardware
//
#include <hip/hip_runtime.h>

// Problem constants (match the reference)
#define BB 4
#define NN 512
#define DD 64
#define HH 8
#define KK 16
#define NBINS 33                 // 2K+1
#define FW   (HH*NBINS + HH*DD)  // 776 feature width
#define A_OFF ((size_t)BB*NN*DD) // embd elements before 'a' in d_out

typedef __attribute__((ext_vector_type(16))) __bf16 v16bf;
typedef __attribute__((ext_vector_type(8)))  __bf16 v8bf;
typedef __attribute__((ext_vector_type(8)))  float  v8f;

static __device__ __forceinline__ __bf16 f2bf(float x) {
    union { float f; unsigned u; } v; v.f = x;
    unsigned r = (v.u + 0x7FFFu + ((v.u >> 16) & 1u)) >> 16;   // RNE truncate
    union { unsigned short s; __bf16 b; } o; o.s = (unsigned short)r;
    return o.b;
}

static __device__ __forceinline__ v16bf cat16(v8bf lo, v8bf hi) {
    return __builtin_shufflevector(lo, hi, 0, 1, 2, 3, 4, 5, 6, 7,
                                            8, 9, 10, 11, 12, 13, 14, 15);
}

// ---------------------------------------------------------------- rank/cumsum
__global__ void rank_kernel(const unsigned char* __restrict__ masks,
                            int* __restrict__ rank, float* __restrict__ mf) {
    int b = blockIdx.x;
    if (threadIdx.x == 0) {
        int c = 0;
        for (int n = 0; n < NN; ++n) {
            int mm = masks[b * NN + n] ? 1 : 0;
            c += mm;
            rank[b * NN + n] = c - 1;
            mf[b * NN + n]   = (float)mm;
        }
    }
}

// ---------------------------------------------------------------- q = s @ Wq^T
// Stored directly in bf16, twice: qbf [B,H,N,D] and qbfT [B,H,D,N].
__global__ void qproj_kernel(const float* __restrict__ s,
                             const float* __restrict__ Wq,
                             __bf16* __restrict__ qbf,
                             __bf16* __restrict__ qbfT) {
    int idx = blockIdx.x * blockDim.x + threadIdx.x;      // ((b*H+h)*N+i)*D+dd
    int dd = idx % DD;
    int i  = (idx / DD) % NN;
    int h  = (idx / (DD * NN)) % HH;
    int b  =  idx / (DD * NN * HH);
    const float* srow = s  + ((size_t)b * NN + i) * DD;
    const float* wrow = Wq + ((size_t)dd * HH + h) * DD;
    float acc = 0.f;
#pragma unroll 8
    for (int c = 0; c < DD; ++c) acc += srow[c] * wrow[c];
    __bf16 v = f2bf(acc);
    qbf[idx] = v;
    qbfT[((size_t)(b * HH + h) * DD + dd) * NN + i] = v;
}

// ---------------------------------------------------------------- attention core
// One wave (32 threads) per (b, h, 16-row i-tile). 1024 blocks total.
#define LSTR 513   // f32 logits LDS row stride
#define PSTR 520   // bf16 prob   LDS row stride (16B aligned, bank-spread)

__global__ void attn_kernel(const __bf16* __restrict__ qbf,   // [B,H,N,D]
                            const __bf16* __restrict__ qbfT,  // [B,H,D,N]
                            const int*    __restrict__ rank,  // [B,N]
                            const float*  __restrict__ mf,    // [B,N]
                            const float*  __restrict__ Wb,    // [H,NB]
                            float* __restrict__ a_out,        // [B,H,N,N]
                            float* __restrict__ feats) {      // [B,N,FW]
    __shared__ float  l_logits[16 * LSTR];
    __shared__ __bf16 l_pbf[16 * PSTR];
    __shared__ float  l_opair[16 * NBINS];
    __shared__ int    l_rank[NN];
    __shared__ float  l_m[NN];
    __shared__ float  l_wb[NBINS];

    const int it = blockIdx.x & 31;
    const int h  = (blockIdx.x >> 5) & (HH - 1);
    const int b  =  blockIdx.x >> 8;
    const int i0 = it * 16;

    const int lane = threadIdx.x;
    const int half = lane >> 4;       // 0|1
    const int lm   = lane & 15;
    const int koff = half << 3;       // 8*(lane>>4): K sub-offset per ISA layout

    // cache per-batch rank/mask rows + bias row, zero o_pair accumulators
    for (int t = lane; t < NN; t += 32) {
        l_rank[t] = rank[b * NN + t];
        l_m[t]    = mf[b * NN + t];
    }
    for (int t = lane; t < NBINS; t += 32) l_wb[t] = Wb[h * NBINS + t];
    for (int t = lane; t < 16 * NBINS; t += 32) l_opair[t] = 0.f;
    __syncthreads();

    const __bf16* qb = qbf  + (size_t)(b * HH + h) * NN * DD;
    const __bf16* qt = qbfT + (size_t)(b * HH + h) * DD * NN;
    const float wL = 0.70710678118654752f;
    const float inv_sqrt_d = 0.125f;  // 1/sqrt(64)

    // A fragments (16x32 bf16): elems 0-7 at K=koff, elems 8-15 at K=16+koff
    v16bf afr[2];
#pragma unroll
    for (int ch = 0; ch < 2; ++ch) {
        const __bf16* base = qb + (size_t)(i0 + lm) * DD + ch * 32 + koff;
        afr[ch] = cat16(*(const v8bf*)base, *(const v8bf*)(base + 16));
    }

    // ---- logits = wL * (q k^T / sqrt(D) + bias) over all 32 j-tiles
    for (int jt = 0; jt < 32; ++jt) {
        int j0 = jt * 16;
        const __bf16* kb = qb + (size_t)(j0 + lm) * DD;           // k = q
        v16bf b0 = cat16(*(const v8bf*)(kb + koff),
                         *(const v8bf*)(kb + 16 + koff));
        v16bf b1 = cat16(*(const v8bf*)(kb + 32 + koff),
                         *(const v8bf*)(kb + 48 + koff));
        if (jt < 31) __builtin_prefetch(kb + 16 * DD, 0, 1);      // next j-tile
        v8f c = {0.f, 0.f, 0.f, 0.f, 0.f, 0.f, 0.f, 0.f};
        c = __builtin_amdgcn_wmma_f32_16x16x32_bf16(false, afr[0], false, b0,
                                                    (short)0, c, false, false);
        c = __builtin_amdgcn_wmma_f32_16x16x32_bf16(false, afr[1], false, b1,
                                                    (short)0, c, false, false);
#pragma unroll
        for (int r = 0; r < 8; ++r) {
            int mi = r + (half << 3);          // C layout: M = r + 8*(lane>>4)
            int i  = i0 + mi;
            int j  = j0 + lm;                  //            N = lane&15
            int dr = l_rank[i] - l_rank[j];
            int bin = min(max(dr, -KK), KK) + KK;
            float sq = l_m[i] * l_m[j];
            l_logits[mi * LSTR + j] = wL * (c[r] * inv_sqrt_d + l_wb[bin] * sq);
        }
    }
    __syncthreads();

    // ---- softmax rows (f32), emit a, store bf16 probs, accumulate o_pair
    for (int r = 0; r < 16; ++r) {
        int i = i0 + r;
        float mx = -1e30f;
        for (int t = lane; t < NN; t += 32) mx = fmaxf(mx, l_logits[r * LSTR + t]);
#pragma unroll
        for (int o = 16; o; o >>= 1) mx = fmaxf(mx, __shfl_xor(mx, o, 32));
        float sm = 0.f;
        for (int t = lane; t < NN; t += 32) sm += __expf(l_logits[r * LSTR + t] - mx);
#pragma unroll
        for (int o = 16; o; o >>= 1) sm += __shfl_xor(sm, o, 32);
        float inv = 1.f / sm;
        float mi_f = l_m[i];
        float* arow = a_out + ((size_t)(b * HH + h) * NN + i) * NN;
        for (int t = lane; t < NN; t += 32) {
            float p  = __expf(l_logits[r * LSTR + t] - mx) * inv;
            float av = p * mi_f * l_m[t];                    // a = sq * softmax
            l_pbf[r * PSTR + t] = f2bf(av);                  // for A@V
            arow[t] = av;
            int dr = l_rank[i] - l_rank[t];
            int bin = min(max(dr, -KK), KK) + KK;
            atomicAdd(&l_opair[r * NBINS + bin], av);        // ds_add_f32
        }
    }
    __syncthreads();

    // ---- o = a @ v  (v = q), bf16 WMMA, K over 512 in chunks of 32
    for (int nt = 0; nt < 4; ++nt) {
        v8f acc = {0.f, 0.f, 0.f, 0.f, 0.f, 0.f, 0.f, 0.f};
        const __bf16* vrow = qt + (size_t)(nt * 16 + lm) * NN;   // v^T row d
        for (int jc = 0; jc < 16; ++jc) {
            const __bf16* pa = &l_pbf[lm * PSTR + jc * 32 + koff];
            v16bf aa = cat16(*(const v8bf*)pa, *(const v8bf*)(pa + 16));
            const __bf16* pb = vrow + jc * 32 + koff;
            v16bf bb = cat16(*(const v8bf*)pb, *(const v8bf*)(pb + 16));
            acc = __builtin_amdgcn_wmma_f32_16x16x32_bf16(false, aa, false, bb,
                                                          (short)0, acc, false, false);
        }
#pragma unroll
        for (int r = 0; r < 8; ++r) {
            int i = i0 + r + (half << 3);
            feats[(size_t)(b * NN + i) * FW + HH * NBINS + h * DD + nt * 16 + lm] = acc[r];
        }
    }

    // ---- o_pair -> feats
    for (int t = lane; t < 16 * NBINS; t += 32) {
        int r = t / NBINS, c = t % NBINS;
        feats[(size_t)(b * NN + i0 + r) * FW + h * NBINS + c] = l_opair[t];
    }
}

// ---------------------------------------------------------------- embd = feats @ Wout^T + bout
__global__ void outproj_kernel(const float* __restrict__ feats,
                               const float* __restrict__ Wout,
                               const float* __restrict__ bout,
                               float* __restrict__ embd) {
    int idx = blockIdx.x * blockDim.x + threadIdx.x;   // (b*N+i)*D + dd
    int dd = idx % DD;
    int bi = idx / DD;
    const float* f = feats + (size_t)bi * FW;
    const float* w = Wout + (size_t)dd * FW;
    float acc = bout[dd];
#pragma unroll 8
    for (int c = 0; c < FW; ++c) acc += f[c] * w[c];
    embd[idx] = acc;
}

// ---------------------------------------------------------------- launcher
extern "C" void kernel_launch(void* const* d_in, const int* in_sizes, int n_in,
                              void* d_out, int out_size, void* d_ws, size_t ws_size,
                              hipStream_t stream) {
    const float*         s     = (const float*)d_in[0];
    const unsigned char* masks = (const unsigned char*)d_in[1];
    const float*         Wq    = (const float*)d_in[2];
    const float*         Wb    = (const float*)d_in[3];
    const float*         Wout  = (const float*)d_in[4];
    const float*         bout  = (const float*)d_in[5];

    // workspace carve-up
    char* ws = (char*)d_ws;
    int*    rank  = (int*)ws;    ws += (size_t)BB * NN * sizeof(int);
    float*  mf    = (float*)ws;  ws += (size_t)BB * NN * sizeof(float);
    __bf16* qbf   = (__bf16*)ws; ws += (size_t)BB * HH * NN * DD * sizeof(__bf16);
    __bf16* qbfT  = (__bf16*)ws; ws += (size_t)BB * HH * NN * DD * sizeof(__bf16);
    float*  feats = (float*)ws;

    float* embd  = (float*)d_out;
    float* a_out = (float*)d_out + A_OFF;

    rank_kernel<<<BB, 32, 0, stream>>>(masks, rank, mf);

    qproj_kernel<<<(BB * HH * NN * DD) / 256, 256, 0, stream>>>(s, Wq, qbf, qbfT);

    attn_kernel<<<BB * HH * (NN / 16), 32, 0, stream>>>(qbf, qbfT, rank, mf, Wb,
                                                        a_out, feats);

    outproj_kernel<<<(BB * NN * DD) / 256, 256, 0, stream>>>(feats, Wout, bout, embd);
}
